// SigModuleParallel_cheng_1614907703714
// MI455X (gfx1250) — compile-verified, matrix operand order checked by reading, not run on hardware
//
#include <hip/hip_runtime.h>
#include <hip/hip_bf16.h>
#include <stdint.h>

// Problem constants (from reference)
#define NB    8      // N
#define CCH   12     // C
#define JJ    25     // J
#define TT    64     // T
#define LL    5      // L
#define D     13     // C + time
#define D2    169
#define D3    2197
#define SIGC  2379   // D + D2 + D3
#define KP    2400   // SIGC padded to multiple of 32 (75 * 32)
#define BROWS 12800  // J*T*N
#define ODIM  256

typedef __attribute__((ext_vector_type(16))) __bf16        v16bf;
typedef __attribute__((ext_vector_type(8)))  float         v8f;
typedef __attribute__((ext_vector_type(4)))  unsigned int  u32x4;

__device__ __forceinline__ unsigned short f2bf(float f) {
    // round-to-nearest-even fp32 -> bf16
    unsigned int u = __float_as_uint(f);
    unsigned int r = u + 0x7FFFu + ((u >> 16) & 1u);
    return (unsigned short)(r >> 16);
}

// ---------------------------------------------------------------------------
// Kernel 1: W [256 x 2379] fp32 -> Wb [256 x 2400] bf16 (zero padded K tail)
// ---------------------------------------------------------------------------
__global__ void wconvert_kernel(const float* __restrict__ W,
                                unsigned short* __restrict__ Wb) {
    int idx = blockIdx.x * blockDim.x + threadIdx.x;  // 0 .. 256*2400-1
    int row = idx / KP;
    int k   = idx - row * KP;
    float v = (k < SIGC) ? W[row * SIGC + k] : 0.0f;
    Wb[idx] = f2bf(v);
}

// ---------------------------------------------------------------------------
// Kernel 2: depth-3 path signature, one wave32 per window.
// s1 (13) + s2 (169) + dx (13) live in per-wave LDS; s3 (2197) is striped
// 69-per-lane in VGPRs. Output row written as bf16 [KP] (padded with zeros).
// ---------------------------------------------------------------------------
__global__ void __launch_bounds__(256) sig_kernel(const float* __restrict__ x,
                                                  unsigned short* __restrict__ sig) {
    __shared__ float sh[8][224];   // per wave: [0..12]=dx, [16..28]=s1, [32..200]=s2
    const int wave = threadIdx.x >> 5;
    const int lane = threadIdx.x & 31;
    const int b    = blockIdx.x * 8 + wave;

    float* dx = &sh[wave][0];
    float* s1 = &sh[wave][16];
    float* s2 = &sh[wave][32];

    // b = (i*T + j)*N + n
    const int n  = b & 7;
    const int rm = b >> 3;
    const int jj = rm & 63;
    const int ii = rm >> 6;

    if (lane < D) s1[lane] = 0.0f;
#pragma unroll
    for (int q = 0; q < 6; ++q) { int m = lane + 32 * q; if (m < D2) s2[m] = 0.0f; }

    float r3[69];
#pragma unroll
    for (int q = 0; q < 69; ++q) r3[q] = 0.0f;

    float prev = 0.0f;                 // zero basepoint
    const float inv6 = 1.0f / 6.0f;
    __syncthreads();

    for (int l = 0; l < LL; ++l) {
        if (lane < D) {
            float cur;
            if (lane < CCH) {
                size_t xi = ((((size_t)n * CCH + lane) * JJ + ii) * TT + jj) * LL + l;
                cur = x[xi];
            } else {
                cur = 0.25f * (float)l;    // time channel linspace(0,1,5)
            }
            dx[lane] = cur - prev;
            prev = cur;
        }
        __syncthreads();

        // s3 += s2 (x) dx + s1 (x) e2 + e3   (reads OLD s1, s2)
#pragma unroll
        for (int q = 0; q < 69; ++q) {
            int m = lane + 32 * q;
            if (m < D3) {
                int i3 = m / D2;
                int r2 = m - i3 * D2;
                int j3 = r2 / D;
                int k3 = r2 - j3 * D;
                float dk = dx[k3], dj = dx[j3], di = dx[i3];
                r3[q] += s2[i3 * D + j3] * dk
                       + s1[i3] * (0.5f * dj * dk)
                       + di * dj * dk * inv6;
            }
        }
        __syncthreads();

        // s2 += s1 (x) dx + e2   (reads OLD s1)
#pragma unroll
        for (int q = 0; q < 6; ++q) {
            int m = lane + 32 * q;
            if (m < D2) {
                int i2 = m / D;
                int j2 = m - i2 * D;
                s2[m] += s1[i2] * dx[j2] + 0.5f * dx[i2] * dx[j2];
            }
        }
        __syncthreads();

        if (lane < D) s1[lane] += dx[lane];
        __syncthreads();
    }

    // emit bf16 row: [s1 | s2 | s3 | zero pad]
    unsigned short* row = sig + (size_t)b * KP;
    if (lane < D) row[lane] = f2bf(s1[lane]);
#pragma unroll
    for (int q = 0; q < 6; ++q) { int m = lane + 32 * q; if (m < D2) row[D + m] = f2bf(s2[m]); }
#pragma unroll
    for (int q = 0; q < 69; ++q) { int m = lane + 32 * q; if (m < D3) row[D + D2 + m] = f2bf(r3[q]); }
    if (lane < (KP - SIGC)) row[SIGC + lane] = 0;  // 21 pad entries
}

// ---------------------------------------------------------------------------
// Kernel 3: WMMA GEMM  out[12800 x 256] = sig[12800 x 2400] * W^T + b,
// scattered into [N, 256, J, T]. One wave computes a 16x64 tile (4 N-subtiles
// reuse one A fragment per 32-wide K step). bf16 A/B, f32 accumulate.
// Fragment layout per CDNA5 ISA (16-bit A 16x32 / B 32x16):
//   lane<16  holds K {kk+0..7, kk+16..23}
//   lane>=16 holds K {kk+8..15, kk+24..31}
// ---------------------------------------------------------------------------
__global__ void __launch_bounds__(256) gemm_kernel(const unsigned short* __restrict__ sigU,
                                                   const unsigned short* __restrict__ WbU,
                                                   const float* __restrict__ bias,
                                                   float* __restrict__ out) {
    const int lane  = threadIdx.x & 31;
    const int wvblk = threadIdx.x >> 5;
    const int wid   = blockIdx.x * 8 + wvblk;   // 0..3199
    const int mt    = wid >> 2;                  // M tile 0..799
    const int ng    = wid & 3;                   // 64-wide N group 0..3
    const int lhalf = lane & 15;
    const int hi    = lane >> 4;

    const __bf16* sig = (const __bf16*)sigU;
    const __bf16* Wb  = (const __bf16*)WbU;

    const __bf16* aRow    = sig + (size_t)(mt * 16 + lhalf) * KP + hi * 8;
    const int     colBase = ng * 64 + lhalf;

    // C/D layout keeps the column fixed per lane -> init accumulator = bias
    v8f acc[4];
#pragma unroll
    for (int nt = 0; nt < 4; ++nt) {
        float bv = bias[colBase + nt * 16];
#pragma unroll
        for (int r = 0; r < 8; ++r) acc[nt][r] = bv;
    }

    for (int kk = 0; kk < KP; kk += 32) {
        union { v16bf v; u32x4 q[2]; } A;
        A.q[0] = *(const u32x4*)(aRow + kk);        // K kk + hi*8 .. +7
        A.q[1] = *(const u32x4*)(aRow + kk + 16);   // K kk+16 + hi*8 .. +7
#pragma unroll
        for (int nt = 0; nt < 4; ++nt) {
            const __bf16* bRow = Wb + (size_t)(colBase + nt * 16) * KP + hi * 8;
            union { v16bf v; u32x4 q[2]; } Bm;
            Bm.q[0] = *(const u32x4*)(bRow + kk);
            Bm.q[1] = *(const u32x4*)(bRow + kk + 16);
            acc[nt] = __builtin_amdgcn_wmma_f32_16x16x32_bf16(
                false, A.v, false, Bm.v, (short)0, acc[nt], false, false);
        }
    }

    // scatter: row -> (n, i, j); out[n, col, i, j]
#pragma unroll
    for (int nt = 0; nt < 4; ++nt) {
        const int col = colBase + nt * 16;
#pragma unroll
        for (int r = 0; r < 8; ++r) {
            int row = mt * 16 + (hi ? (r + 8) : r);
            int n   = row & 7;
            int rm  = row >> 3;
            int j   = rm & 63;
            int i   = rm >> 6;
            size_t oi = (((size_t)n * ODIM + col) * JJ + i) * TT + j;
            out[oi] = acc[nt][r];
        }
    }
}

// ---------------------------------------------------------------------------
extern "C" void kernel_launch(void* const* d_in, const int* in_sizes, int n_in,
                              void* d_out, int out_size, void* d_ws, size_t ws_size,
                              hipStream_t stream) {
    const float* x = (const float*)d_in[0];   // [8,12,25,64,5]
    const float* W = (const float*)d_in[1];   // [256,2379]
    const float* b = (const float*)d_in[2];   // [256]
    float* out = (float*)d_out;               // [8,256,25,64]

    unsigned short* sig = (unsigned short*)d_ws;            // 12800*2400*2 = 61,440,000 B
    unsigned short* Wb  = sig + (size_t)BROWS * KP;         // + 1,228,800 B

    wconvert_kernel<<<(ODIM * KP) / 256, 256, 0, stream>>>(W, Wb);
    sig_kernel<<<BROWS / 8, 256, 0, stream>>>(x, sig);
    gemm_kernel<<<(BROWS / 16) * 4 / 8, 256, 0, stream>>>(sig, Wb, b, out);
}
